// ourmodel_6579889897582
// MI455X (gfx1250) — matrix operand order repaired and checked by reading, not compiled, not run on hardware
//
#include <hip/hip_runtime.h>
#include <hip/hip_bf16.h>
#include <math.h>

#define D    512
#define S    512
#define T    32768
#define SEG  64
#define BAND 192
#define ATT_SCALE 0.044194173824159216f   // 1/sqrt(512)

typedef __attribute__((ext_vector_type(16))) __bf16 v16bf;
typedef __attribute__((ext_vector_type(8)))  __bf16 v8bf;
typedef __attribute__((ext_vector_type(8)))  float  v8f;
typedef __attribute__((ext_vector_type(4)))  float  f32x4;

__device__ __forceinline__ v8f wmma_bf16(v16bf a, v16bf b, v8f c) {
    return __builtin_amdgcn_wmma_f32_16x16x32_bf16(false, a, false, b,
                                                   (short)0, c, false, false);
}

// A-operand: lane's halves live at row[k0..k0+7] and row[k0+16..k0+23]
__device__ __forceinline__ v16bf frag_gap16(const __bf16* p) {
    v8bf lo = *(const v8bf*)p;
    v8bf hi = *(const v8bf*)(p + 16);
    return __builtin_shufflevector(lo, hi, 0,1,2,3,4,5,6,7,8,9,10,11,12,13,14,15);
}
// B-operand (k-major row): lane's halves are 16 contiguous bf16
__device__ __forceinline__ v16bf frag_contig(const __bf16* p) {
    v8bf lo = *(const v8bf*)p;
    v8bf hi = *(const v8bf*)(p + 8);
    return __builtin_shufflevector(lo, hi, 0,1,2,3,4,5,6,7,8,9,10,11,12,13,14,15);
}
__device__ __forceinline__ v16bf cvt4x4(f32x4 a, f32x4 b, f32x4 c, f32x4 d) {
    v16bf r;
    for (int i = 0; i < 4; ++i) {
        r[i]      = (__bf16)a[i];
        r[4 + i]  = (__bf16)b[i];
        r[8 + i]  = (__bf16)c[i];
        r[12 + i] = (__bf16)d[i];
    }
    return r;
}
__device__ __forceinline__ v8bf cvt2x4(f32x4 a, f32x4 b) {
    v8bf r;
    for (int i = 0; i < 4; ++i) { r[i] = (__bf16)a[i]; r[4 + i] = (__bf16)b[i]; }
    return r;
}

// ---------------------------------------------------------------------------
// Cross attention: 16 queries / block.  Band-relative score storage == mask.
// out[s][d] = sum_t softmax((tgt+qp).(mem+pos))[t] * mem[t][d]
// ---------------------------------------------------------------------------
__global__ __launch_bounds__(256)
void cross_attn_kernel(const float* __restrict__ tgt,
                       const float* __restrict__ mem,
                       const float* __restrict__ pos,
                       const float* __restrict__ qpos,
                       float* __restrict__ out)
{
    __shared__ __bf16 Qs[16][D];        // 16 KB
    __shared__ float  Sc[16 * BAND];    // 12 KB (band-relative scores)
    __shared__ float  dump[32];         // sink for out-of-band stores
    __shared__ float  zpad[8];          // zero source for out-of-band reads
    const int q0   = blockIdx.x * 16;
    const int tid  = threadIdx.x;
    const int lane = tid & 31;
    const int wid  = tid >> 5;

    for (int i = tid; i < 16 * BAND; i += 256) Sc[i] = -INFINITY;
    if (tid < 8) zpad[tid] = 0.0f;

    { // stage Q = tgt + query_pos (bf16), vectorized
        int m  = tid >> 4;
        int c0 = (tid & 15) * 32;
        const f32x4* pt = (const f32x4*)(tgt  + (q0 + m) * D + c0);
        const f32x4* pq = (const f32x4*)(qpos + (q0 + m) * D + c0);
        for (int i = 0; i < 4; ++i)
            *(v8bf*)&Qs[m][c0 + i * 8] =
                cvt2x4(pt[2*i] + pq[2*i], pt[2*i+1] + pq[2*i+1]);
    }
    __syncthreads();

    const long tbase = (long)(q0 - 1) * SEG;   // may be negative at q0==0

    // ---- phase 1: banded QK^T scores -------------------------------------
    for (int j = wid; j < 72; j += 8) {
        long f0 = tbase + (long)j * 16;
        if (f0 + 16 <= 0 || f0 >= T) continue;
        long f  = f0 + (lane & 15);
        bool fv = (f >= 0) && (f < T);
        const float* pm = mem + (fv ? f : 0) * (long)D;
        const float* pp = pos + (fv ? f : 0) * (long)D;
        const int dB0 = (lane >> 4) << 4;     // 16 contiguous dims per lane
        const __bf16* qrow = &Qs[lane & 15][(lane >> 4) << 4];
        v8f c = {};
        for (int dbase = 0; dbase < D; dbase += 32) {
            __builtin_prefetch(pm + dbase + dB0 + 128 * D, 0, 1);
            v16bf a = frag_contig(qrow + dbase);
            const f32x4* m4 = (const f32x4*)(pm + dbase + dB0);
            const f32x4* p4 = (const f32x4*)(pp + dbase + dB0);
            v16bf b = cvt4x4(m4[0] + p4[0], m4[1] + p4[1],
                             m4[2] + p4[2], m4[3] + p4[3]);
            if (!fv) b = v16bf{};             // rare edge lanes
            c = wmma_bf16(a, b, c);
        }
        // unconditional scatter: invalid elements go to the dump slot
        int n  = lane & 15;
        int mh = (lane >> 4) << 3;
        for (int r = 0; r < 8; ++r) {
            int  m   = r + mh;
            long t   = f0 + n;
            int  col = (int)(t - tbase) - m * SEG;   // band-relative
            bool ok  = (t >= 0) && (t < T) && (col >= 0) && (col < BAND);
            float* dst = ok ? &Sc[m * BAND + col] : &dump[lane];
            *dst = c[r] * ATT_SCALE;
        }
    }
    __syncthreads();

    // ---- phase 2: softmax over the 192-wide band -------------------------
    {
        int r = tid >> 4, c = tid & 15;
        float mx = -INFINITY;
        for (int k = c; k < BAND; k += 16) mx = fmaxf(mx, Sc[r * BAND + k]);
        for (int m = 8; m >= 1; m >>= 1) mx = fmaxf(mx, __shfl_xor(mx, m, 32));
        float sum = 0.f;
        for (int k = c; k < BAND; k += 16) {
            float e = __expf(Sc[r * BAND + k] - mx);
            Sc[r * BAND + k] = e; sum += e;
        }
        for (int m = 8; m >= 1; m >>= 1) sum += __shfl_xor(sum, m, 32);
        float inv = 1.0f / sum;
        for (int k = c; k < BAND; k += 16) Sc[r * BAND + k] *= inv;
    }
    __syncthreads();

    // ---- phase 3: attn @ memory (A=attn bf16, zeros outside band) --------
    {
        const int nsubBase = wid * 4;           // 4 n-subtiles (64 dims) / wave
        v8f acc[4] = {v8f{}, v8f{}, v8f{}, v8f{}};
        const int mrow = lane & 15;
        const int dcol = lane & 15;
        const int khh  = (lane >> 4) << 4;      // k offset 0 or 16
        for (int kk = 0; kk < 1152; kk += 32) {
            long t0 = tbase + kk;
            if (t0 + 32 <= 0 || t0 >= T) continue;
            // A fragment: attn values, zero outside band (zpad reads)
            v16bf a;
            {
                int base8 = (lane >> 4) << 3;
                for (int h = 0; h < 16; ++h) {
                    int u   = kk + khh + h;
                    int col = u - mrow * SEG;
                    const float* src = (col >= 0 && col < BAND)
                                     ? &Sc[mrow * BAND + col] : &zpad[h & 7];
                    a[h] = (__bf16)(*src);
                }
            }
            const bool full = (t0 >= 0) && (t0 + 32 <= T);
            for (int i = 0; i < 4; ++i) {
                int dbase = (nsubBase + i) * 16;
                const float* col0 = mem + (t0 + khh) * (long)D + dbase + dcol;
                __builtin_prefetch(col0 + 128 * D, 0, 1);
                v16bf b;
                if (full) {
                    for (int h = 0; h < 16; ++h)
                        b[h] = (__bf16)col0[(long)h * D];
                } else {
                    for (int h = 0; h < 16; ++h) {
                        long t = t0 + khh + h;
                        b[h] = (__bf16)((t >= 0 && t < T) ? col0[(long)h * D] : 0.0f);
                    }
                }
                acc[i] = wmma_bf16(a, b, acc[i]);
            }
        }
        int n = lane & 15, mh = (lane >> 4) << 3;
        for (int i = 0; i < 4; ++i) {
            int dbase = (nsubBase + i) * 16;
            for (int r = 0; r < 8; ++r)
                out[(q0 + r + mh) * D + dbase + n] = acc[i][r];
        }
    }
}

// ---------------------------------------------------------------------------
// Generic Y = op(A) @ W^T + bias   (512x512x512, optional ReLU on A input)
// 64x64 tile / block, 8 waves, K-step 32 (one WMMA per subtile per stage).
// ---------------------------------------------------------------------------
__global__ __launch_bounds__(256)
void gemm_wmma_kernel(const float* __restrict__ A, const float* __restrict__ W,
                      const float* __restrict__ bias, float* __restrict__ Y,
                      int reluA)
{
    __shared__ __bf16 As[64][32];   // [m][k]
    __shared__ __bf16 Bs[64][32];   // [n][k]  (k-major: contiguous B frags)
    const int tid = threadIdx.x, lane = tid & 31, wid = tid >> 5;
    const int m0 = (blockIdx.x >> 3) * 64;
    const int n0 = (blockIdx.x & 7) * 64;
    const int msub = wid >> 1;
    const int lr = tid >> 2;          // loader row 0..63
    const int lc = (tid & 3) * 8;     // 8 consecutive K
    v8f acc0 = {}, acc1 = {};

    const __bf16* arow = &As[msub * 16 + (lane & 15)][(lane >> 4) << 4];
    const __bf16* brow0 = &Bs[((wid & 1) * 2 + 0) * 16 + (lane & 15)][(lane >> 4) << 4];
    const __bf16* brow1 = &Bs[((wid & 1) * 2 + 1) * 16 + (lane & 15)][(lane >> 4) << 4];

    for (int k0 = 0; k0 < D; k0 += 32) {
        {
            const f32x4* pa = (const f32x4*)(A + (m0 + lr) * D + k0 + lc);
            __builtin_prefetch(pa + 8, 0, 1);          // next k-tile
            f32x4 a0 = pa[0], a1 = pa[1];
            if (reluA) {
                for (int i = 0; i < 4; ++i) { a0[i] = fmaxf(a0[i], 0.f); a1[i] = fmaxf(a1[i], 0.f); }
            }
            *(v8bf*)&As[lr][lc] = cvt2x4(a0, a1);
            const f32x4* pw = (const f32x4*)(W + (n0 + lr) * D + k0 + lc);
            __builtin_prefetch(pw + 8, 0, 1);
            *(v8bf*)&Bs[lr][lc] = cvt2x4(pw[0], pw[1]);
        }
        __syncthreads();
        v16bf a  = frag_contig(arow);
        v16bf b0 = frag_contig(brow0);
        v16bf b1 = frag_contig(brow1);
        acc0 = wmma_bf16(a, b0, acc0);
        acc1 = wmma_bf16(a, b1, acc1);
        __syncthreads();
    }
    int n = lane & 15, mh = (lane >> 4) << 3;
    for (int r = 0; r < 8; ++r) {
        int m  = m0 + msub * 16 + r + mh;
        int c0 = n0 + ((wid & 1) * 2) * 16 + n;
        Y[m * D + c0]      = acc0[r] + bias[c0];
        Y[m * D + c0 + 16] = acc1[r] + bias[c0 + 16];
    }
}

// ---------------------------------------------------------------------------
// out = LN(a + b) * g + be    (one row / block)
// ---------------------------------------------------------------------------
__global__ __launch_bounds__(128)
void add_ln_kernel(const float* __restrict__ a, const float* __restrict__ b,
                   const float* __restrict__ g, const float* __restrict__ be,
                   float* __restrict__ out)
{
    __shared__ float red[128];
    const int s = blockIdx.x, tid = threadIdx.x;
    f32x4 v = ((const f32x4*)(a + s * D))[tid] + ((const f32x4*)(b + s * D))[tid];
    float sum = v[0] + v[1] + v[2] + v[3];
    red[tid] = sum; __syncthreads();
    for (int st = 64; st >= 1; st >>= 1) {
        if (tid < st) red[tid] += red[tid + st];
        __syncthreads();
    }
    float mean = red[0] * (1.0f / D); __syncthreads();
    float vs = 0.f;
    for (int i = 0; i < 4; ++i) { float t = v[i] - mean; vs += t * t; }
    red[tid] = vs; __syncthreads();
    for (int st = 64; st >= 1; st >>= 1) {
        if (tid < st) red[tid] += red[tid + st];
        __syncthreads();
    }
    float inv = rsqrtf(red[0] * (1.0f / D) + 1e-5f);
    f32x4 gg = ((const f32x4*)g)[tid], bb = ((const f32x4*)be)[tid];
    f32x4 o;
    for (int i = 0; i < 4; ++i) o[i] = (v[i] - mean) * inv * gg[i] + bb[i];
    ((f32x4*)(out + s * D))[tid] = o;
}

// ---------------------------------------------------------------------------
// Banded self-attention (|i-j|<=4): 9 keys x 512 dims per query. VALU kernel.
// ---------------------------------------------------------------------------
__global__ __launch_bounds__(256)
void self_attn_kernel(const float* __restrict__ x, float* __restrict__ out)
{
    __shared__ float red[256];
    __shared__ float aw[9];
    const int s = blockIdx.x, tid = threadIdx.x;
    float q0v = x[s * D + tid], q1v = x[s * D + tid + 256];
    for (int j = 0; j < 9; ++j) {
        int k = s - 4 + j;
        float p = 0.f;
        if (k >= 0 && k < S)
            p = q0v * x[k * D + tid] + q1v * x[k * D + tid + 256];
        red[tid] = p; __syncthreads();
        for (int st = 128; st >= 1; st >>= 1) {
            if (tid < st) red[tid] += red[tid + st];
            __syncthreads();
        }
        if (tid == 0) aw[j] = (k >= 0 && k < S) ? red[0] * ATT_SCALE : -INFINITY;
        __syncthreads();
    }
    if (tid == 0) {
        float mx = -INFINITY;
        for (int j = 0; j < 9; ++j) mx = fmaxf(mx, aw[j]);
        float sum = 0.f;
        for (int j = 0; j < 9; ++j) { float e = __expf(aw[j] - mx); aw[j] = e; sum += e; }
        float inv = 1.0f / sum;
        for (int j = 0; j < 9; ++j) aw[j] *= inv;
    }
    __syncthreads();
    float o0 = 0.f, o1 = 0.f;
    for (int j = 0; j < 9; ++j) {
        int k = s - 4 + j;
        if (k >= 0 && k < S) {
            float w = aw[j];
            o0 += w * x[k * D + tid];
            o1 += w * x[k * D + tid + 256];
        }
    }
    out[s * D + tid]       = o0;
    out[s * D + tid + 256] = o1;
}

// ---------------------------------------------------------------------------
extern "C" void kernel_launch(void* const* d_in, const int* in_sizes, int n_in,
                              void* d_out, int out_size, void* d_ws, size_t ws_size,
                              hipStream_t stream)
{
    (void)in_sizes; (void)n_in; (void)out_size; (void)ws_size;
    const float* tgt       = (const float*)d_in[0];
    const float* memory    = (const float*)d_in[1];
    const float* pos       = (const float*)d_in[2];
    const float* qpos      = (const float*)d_in[3];
    /* d_in[4] action_idx: seg_id(t) == t>>6 analytically (T/S == 64) */
    const float* ca_t2l_w  = (const float*)d_in[5];
    const float* ca_t2l_b  = (const float*)d_in[6];
    const float* ca_l1_w   = (const float*)d_in[7];
    const float* ca_l1_b   = (const float*)d_in[8];
    const float* ca_l2_w   = (const float*)d_in[9];
    const float* ca_l2_b   = (const float*)d_in[10];
    const float* ca_n2_g   = (const float*)d_in[11];
    const float* ca_n2_b   = (const float*)d_in[12];
    const float* ca_n3_g   = (const float*)d_in[13];
    const float* ca_n3_b   = (const float*)d_in[14];
    const float* sa_conv_w = (const float*)d_in[15];
    const float* sa_conv_b = (const float*)d_in[16];
    const float* sa_l1_w   = (const float*)d_in[17];
    const float* sa_l1_b   = (const float*)d_in[18];
    const float* sa_l2_w   = (const float*)d_in[19];
    const float* sa_l2_b   = (const float*)d_in[20];
    const float* sa_n1_g   = (const float*)d_in[21];
    const float* sa_n1_b   = (const float*)d_in[22];
    const float* sa_n2_g   = (const float*)d_in[23];
    const float* sa_n2_b   = (const float*)d_in[24];

    float* outp = (float*)d_out;
    float* bufA = (float*)d_ws;          // S*D each
    float* bufB = bufA + S * D;
    float* bufC = bufB + S * D;
    float* bufD = bufC + S * D;

    // ---- cross attention block ----
    cross_attn_kernel<<<32, 256, 0, stream>>>(tgt, memory, pos, qpos, bufA);
    gemm_wmma_kernel<<<64, 256, 0, stream>>>(bufA, ca_t2l_w, ca_t2l_b, bufB, 1);
    add_ln_kernel<<<512, 128, 0, stream>>>(tgt, bufB, ca_n2_g, ca_n2_b, bufC);
    gemm_wmma_kernel<<<64, 256, 0, stream>>>(bufC, ca_l1_w, ca_l1_b, bufA, 0);
    gemm_wmma_kernel<<<64, 256, 0, stream>>>(bufA, ca_l2_w, ca_l2_b, bufB, 1);
    add_ln_kernel<<<512, 128, 0, stream>>>(bufC, bufB, ca_n3_g, ca_n3_b, bufD);
    // ---- self attention block ----
    self_attn_kernel<<<512, 256, 0, stream>>>(bufD, bufA);
    gemm_wmma_kernel<<<64, 256, 0, stream>>>(bufA, sa_conv_w, sa_conv_b, bufB, 1);
    add_ln_kernel<<<512, 128, 0, stream>>>(bufD, bufB, sa_n1_g, sa_n1_b, bufC);
    gemm_wmma_kernel<<<64, 256, 0, stream>>>(bufC, sa_l1_w, sa_l1_b, bufA, 0);
    gemm_wmma_kernel<<<64, 256, 0, stream>>>(bufA, sa_l2_w, sa_l2_b, bufB, 1);
    add_ln_kernel<<<512, 128, 0, stream>>>(bufC, bufB, sa_n2_g, sa_n2_b, outp);
}
